// GRUModule_1623497638297
// MI455X (gfx1250) — matrix operand order repaired and checked by reading, not compiled, not run on hardware
//
#include <hip/hip_runtime.h>

typedef __attribute__((ext_vector_type(8)))  __bf16 v8bf;
typedef __attribute__((ext_vector_type(16))) __bf16 v16bf;
typedef __attribute__((ext_vector_type(8)))  float  v8f;
typedef __attribute__((ext_vector_type(4)))  float  v4f;

#define DEVFN static __device__ __forceinline__

static constexpr int IN_DIM = 64, H = 128, OUT_DIM = 64;
static constexpr int Bsz = 1024, Tlen = 512;
static constexpr int H3 = 3 * H;
static constexpr float NEG_SLOPE = 0.01f;
static constexpr float LN_EPS = 1e-5f;

// LDS row strides (elements), padded to rotate banks
static constexpr int WSTR  = H + 8;        // 136 bf16 -> 272B rows, 4-bank rotation
static constexpr int W1STR = IN_DIM + 8;   // 72 bf16
static constexpr int HSTR  = H + 4;        // 132 f32 -> 528B rows
static constexpr int HBSTR = H + 8;        // 136 bf16 shadow-state stride
static constexpr int RZSTR = 2 * H + 4;    // 260 f32

DEVFN v16bf cat16(v8bf lo, v8bf hi) {
  v16bf r;
#pragma unroll
  for (int i = 0; i < 8; ++i) { r[i] = lo[i]; r[8 + i] = hi[i]; }
  return r;
}

DEVFN v8bf cvt8(const float* p) {           // 8 consecutive f32 (16B-aligned) -> 8 bf16
  v4f a = *(const v4f*)p;
  v4f b = *(const v4f*)(p + 4);
  v8bf r;
#pragma unroll
  for (int i = 0; i < 4; ++i) { r[i] = (__bf16)a[i]; r[4 + i] = (__bf16)b[i]; }
  return r;
}

// A fragment (16xK tile, chunk c -> K=32c..32c+31) from row-major f32, row stride S.
// ISA 7.12.2 16-bit A 16x32: lanes 0-15 rows M=lane hold K 0-7,16-23; lanes 16-31 K 8-15,24-31.
DEVFN v16bf afrag_f32(const float* base, int S, int c, int lane) {
  const int hi = lane >> 4, m = lane & 15;
  const float* r = base + m * S + c * 32;
  return cat16(cvt8(r + 8 * hi), cvt8(r + 16 + 8 * hi));
}

// Same fragment from row-major bf16 (global or LDS), row stride S (elements).
DEVFN v16bf afrag_bf16(const __bf16* base, long S, int c, int lane) {
  const int hi = lane >> 4, m = lane & 15;
  const __bf16* r = base + (long)m * S + c * 32;
  v8bf lo = *(const v8bf*)(r + 8 * hi);
  v8bf hh = *(const v8bf*)(r + 16 + 8 * hi);
  return cat16(lo, hh);
}

// B fragment for out = X @ W.T : B(k,n) = W[n][k]. W row-major [N][K] bf16, stride S.
// lanes 0-15 col N=lane (K=32c+0..7,16..23), lanes 16-31 col N=lane-16 (K=32c+8..15,24..31).
DEVFN v16bf bfrag(const __bf16* W, int S, int nt, int c, int lane) {
  const int hi = lane >> 4;
  const int n = nt * 16 + (lane & 15);
  const __bf16* r = W + n * S + c * 32 + 8 * hi;
  v8bf lo = *(const v8bf*)r;
  v8bf hh = *(const v8bf*)(r + 16);
  return cat16(lo, hh);
}

DEVFN v8f wmma_bf16(v16bf a, v16bf b, v8f c) {
  return __builtin_amdgcn_wmma_f32_16x16x32_bf16(false, a, false, b, (short)0, c,
                                                 false, false);
}

// bias + LeakyReLU + LayerNorm over the 128-wide row held in acc[0..7] C-tiles,
// result written (bf16) to LDS buffer hb [16][HBSTR].
DEVFN void act_ln_store(v8f (&acc)[8], const float* bias, const float* gam,
                        const float* bet, __bf16* hb, int lane) {
  const int hi = lane >> 4, lm = lane & 15;
#pragma unroll
  for (int nt = 0; nt < 8; ++nt) {
    float bs = bias[nt * 16 + lm];
#pragma unroll
    for (int v = 0; v < 8; ++v) {
      float h = acc[nt][v] + bs;
      acc[nt][v] = h > 0.f ? h : NEG_SLOPE * h;
    }
  }
#pragma unroll
  for (int v = 0; v < 8; ++v) {
    float s1 = 0.f, s2 = 0.f;
#pragma unroll
    for (int nt = 0; nt < 8; ++nt) { float h = acc[nt][v]; s1 += h; s2 += h * h; }
#pragma unroll
    for (int msk = 1; msk < 16; msk <<= 1) {   // reduce within 16-lane half
      s1 += __shfl_xor(s1, msk, 32);
      s2 += __shfl_xor(s2, msk, 32);
    }
    const float mu = s1 * (1.f / 128.f);
    const float var = s2 * (1.f / 128.f) - mu * mu;
    const float rs = rsqrtf(var + LN_EPS);
    const int m = v + 8 * hi;
#pragma unroll
    for (int nt = 0; nt < 8; ++nt) {
      const int n = nt * 16 + lm;
      hb[m * HBSTR + n] = (__bf16)(gam[n] * (acc[nt][v] - mu) * rs + bet[n]);
    }
  }
}

// ---------------- Kernel 1: fused MLP (Linear+LeakyReLU+LN) x2, emits h as bf16 ----
__global__ __launch_bounds__(128) void mlp_fused_kernel(
    const float* __restrict__ x,
    const float* __restrict__ W1, const float* __restrict__ b1,
    const float* __restrict__ g1, const float* __restrict__ be1,
    const float* __restrict__ W2, const float* __restrict__ b2,
    const float* __restrict__ g2, const float* __restrict__ be2,
    __bf16* __restrict__ h_ws) {
  __shared__ __bf16 sW1[H * W1STR];
  __shared__ __bf16 sW2[H * WSTR];
  __shared__ float  sP[6 * H];
  __shared__ __bf16 hbuf[4][16 * HBSTR];

  const int tid = threadIdx.x;
  {
    const int r = tid;                       // 128 threads <-> 128 rows
#pragma unroll 4
    for (int k = 0; k < IN_DIM; ++k) sW1[r * W1STR + k] = (__bf16)W1[r * IN_DIM + k];
#pragma unroll 4
    for (int k = 0; k < H; ++k)      sW2[r * WSTR + k]  = (__bf16)W2[r * H + k];
    sP[0 * H + r] = b1[r];  sP[1 * H + r] = g1[r];  sP[2 * H + r] = be1[r];
    sP[3 * H + r] = b2[r];  sP[4 * H + r] = g2[r];  sP[5 * H + r] = be2[r];
  }
  __syncthreads();

  const int wave = tid >> 5, lane = tid & 31;
  __bf16* hb = hbuf[wave];

  for (int tile = 0; tile < 4; ++tile) {
    const long row0 = (long)blockIdx.x * 256 + wave * 64 + tile * 16;

    // ---- layer 1: [16,64] @ W1.T -> [16,128]
    const float* xt = x + row0 * IN_DIM;
    v16bf ax0 = afrag_f32(xt, IN_DIM, 0, lane);
    v16bf ax1 = afrag_f32(xt, IN_DIM, 1, lane);
    v8f acc[8];
#pragma unroll
    for (int nt = 0; nt < 8; ++nt) {
      v8f c = {};
      c = wmma_bf16(ax0, bfrag(sW1, W1STR, nt, 0, lane), c);
      c = wmma_bf16(ax1, bfrag(sW1, W1STR, nt, 1, lane), c);
      acc[nt] = c;
    }
    act_ln_store(acc, sP + 0 * H, sP + 1 * H, sP + 2 * H, hb, lane);

    // ---- layer 2: [16,128] @ W2.T -> [16,128]  (A frags straight from bf16 LDS)
    v16bf a2[4];
#pragma unroll
    for (int c = 0; c < 4; ++c) a2[c] = afrag_bf16(hb, HBSTR, c, lane);
    v8f acc2[8];
#pragma unroll
    for (int nt = 0; nt < 8; ++nt) {
      v8f c2 = {};
#pragma unroll
      for (int c = 0; c < 4; ++c)
        c2 = wmma_bf16(a2[c], bfrag(sW2, WSTR, nt, c, lane), c2);
      acc2[nt] = c2;
    }
    act_ln_store(acc2, sP + 3 * H, sP + 4 * H, sP + 5 * H, hb, lane);

    // ---- emit h tile as bf16, coalesced 16B chunks
#pragma unroll
    for (int i = 0; i < 8; ++i) {
      const int linear = i * 32 + lane;      // 256 chunks of 8 elems
      const int m = linear >> 4;
      const int kb = (linear & 15) * 8;
      v8bf o = *(const v8bf*)(hb + m * HBSTR + kb);
      *(v8bf*)(h_ws + (row0 + m) * (long)H + kb) = o;
    }
  }
}

// ---------------- Kernel 2: GRU scan, gates fused (x-side + h-side GEMMs), out GEMM ----
__global__ __launch_bounds__(32) void gru_scan_kernel(
    const unsigned char* __restrict__ is_init,
    const float* __restrict__ hx,
    const float* __restrict__ Wih, const float* __restrict__ Whh,
    const float* __restrict__ bih, const float* __restrict__ bhh,
    const float* __restrict__ Wout, const float* __restrict__ bout,
    const __bf16* __restrict__ h_ws,
    float* __restrict__ out, float* __restrict__ hxf) {
  __shared__ __bf16 sWih[H3 * WSTR];        // 104 KB
  __shared__ __bf16 sWhh[H3 * WSTR];        // 104 KB
  __shared__ __bf16 sWout[OUT_DIM * WSTR];  // 17 KB
  __shared__ float  sh[16 * HSTR];          // h state tile, f32 master copy
  __shared__ __bf16 shb[16 * HBSTR];        // h state tile, bf16 shadow for A-frags
  __shared__ float  srz[16 * RZSTR];        // r | z
  __shared__ float  sbih[H3], sbhh[H3], sbout[OUT_DIM];

  const int lane = threadIdx.x;
  const int hi = lane >> 4, lm = lane & 15;
  const int b0 = blockIdx.x * 16;

  // one-time: weights -> bf16 LDS (L2-hot, amortized over T=512 steps)
  for (int r = lane; r < H3; r += 32) {
    for (int k = 0; k < H; ++k) {
      sWih[r * WSTR + k] = (__bf16)Wih[r * H + k];
      sWhh[r * WSTR + k] = (__bf16)Whh[r * H + k];
    }
    sbih[r] = bih[r];  sbhh[r] = bhh[r];
  }
  for (int r = lane; r < OUT_DIM; r += 32) {
    for (int k = 0; k < H; ++k) sWout[r * WSTR + k] = (__bf16)Wout[r * H + k];
    sbout[r] = bout[r];
  }
  for (int i = lane; i < 16 * H; i += 32) {
    const int m = i >> 7, k = i & 127;
    const float v = hx[(long)(b0 + m) * H + k];
    sh[m * HSTR + k]  = v;
    shb[m * HBSTR + k] = (__bf16)v;
  }
  __syncthreads();

  for (int t = 0; t < Tlen; ++t) {
    // reset-on-init (both f32 master and bf16 shadow)
    if (lane < 16) {
      if (is_init[(long)(b0 + lane) * Tlen + t]) {
#pragma unroll 4
        for (int k = 0; k < H; ++k) {
          sh[lane * HSTR + k]  = 0.f;
          shb[lane * HBSTR + k] = (__bf16)0.f;
        }
      }
    }
    __syncthreads();

    // A fragments: h_prev (bf16 LDS shadow) and h_mlp[:,t,:] (global bf16)
    v16bf Ah[4], Ax[4];
#pragma unroll
    for (int c = 0; c < 4; ++c) Ah[c] = afrag_bf16(shb, HBSTR, c, lane);
    const __bf16* xrow = h_ws + ((long)b0 * Tlen + t) * H;
#pragma unroll
    for (int c = 0; c < 4; ++c) Ax[c] = afrag_bf16(xrow, (long)Tlen * H, c, lane);

    // prefetch next timestep's h_mlp rows (global_prefetch_b8) behind this step's WMMAs
    if (t + 1 < Tlen) {
      const __bf16* nrow = h_ws + ((long)b0 * Tlen + (t + 1)) * H;
      __builtin_prefetch(nrow + (long)lm * Tlen * H + hi * H, 0, 3);
    }

    // pass A: r and z gates (x-side and h-side contributions combine)
    for (int nt = 0; nt < 16; ++nt) {
      v8f acc = {};
#pragma unroll
      for (int c = 0; c < 4; ++c) {
        acc = wmma_bf16(Ax[c], bfrag(sWih, WSTR, nt, c, lane), acc);
        acc = wmma_bf16(Ah[c], bfrag(sWhh, WSTR, nt, c, lane), acc);
      }
      const int n = nt * 16 + lm;
      const float bs = sbih[n] + sbhh[n];
#pragma unroll
      for (int v = 0; v < 8; ++v)
        srz[(v + 8 * hi) * RZSTR + n] = 1.f / (1.f + __expf(-(acc[v] + bs)));
    }
    __syncthreads();

    // pass B: n gate needs xn and hn separate (r scales only hn), then h update
    for (int j = 0; j < 8; ++j) {
      v8f ax = {}, ah = {};
#pragma unroll
      for (int c = 0; c < 4; ++c) {
        ax = wmma_bf16(Ax[c], bfrag(sWih, WSTR, 16 + j, c, lane), ax);
        ah = wmma_bf16(Ah[c], bfrag(sWhh, WSTR, 16 + j, c, lane), ah);
      }
      const int col = j * 16 + lm;
      const float bx = sbih[2 * H + col], bh = sbhh[2 * H + col];
#pragma unroll
      for (int v = 0; v < 8; ++v) {
        const int m = v + 8 * hi;
        const float r  = srz[m * RZSTR + col];
        const float z  = srz[m * RZSTR + H + col];
        const float nv = tanhf(ax[v] + bx + r * (ah[v] + bh));
        const float hp = sh[m * HSTR + col];
        const float hn = (1.f - z) * nv + z * hp;   // in-place: disjoint cols/tile
        sh[m * HSTR + col]  = hn;
        shb[m * HBSTR + col] = (__bf16)hn;
      }
    }
    __syncthreads();

    // output projection: y_t @ W_out.T -> [16,64]  (A frags from bf16 shadow)
    v16bf Ay[4];
#pragma unroll
    for (int c = 0; c < 4; ++c) Ay[c] = afrag_bf16(shb, HBSTR, c, lane);
#pragma unroll
    for (int ot = 0; ot < 4; ++ot) {
      v8f acc = {};
#pragma unroll
      for (int c = 0; c < 4; ++c)
        acc = wmma_bf16(Ay[c], bfrag(sWout, WSTR, ot, c, lane), acc);
      const int col = ot * 16 + lm;
      const float bo = sbout[col];
#pragma unroll
      for (int v = 0; v < 8; ++v) {
        const int m = v + 8 * hi;
        out[((long)(b0 + m) * Tlen + t) * OUT_DIM + col] = acc[v] + bo;
      }
    }
  }

  // hx_final (full f32 state)
  for (int i = lane; i < 16 * H; i += 32) {
    const int m = i >> 7, k = i & 127;
    hxf[(long)(b0 + m) * H + k] = sh[m * HSTR + k];
  }
}

extern "C" void kernel_launch(void* const* d_in, const int* in_sizes, int n_in,
                              void* d_out, int out_size, void* d_ws, size_t ws_size,
                              hipStream_t stream) {
  const float* x            = (const float*)d_in[0];
  const unsigned char* isin = (const unsigned char*)d_in[1];   // bool
  const float* hx   = (const float*)d_in[2];
  const float* W1   = (const float*)d_in[3];
  const float* b1   = (const float*)d_in[4];
  const float* g1   = (const float*)d_in[5];
  const float* be1  = (const float*)d_in[6];
  const float* W2   = (const float*)d_in[7];
  const float* b2   = (const float*)d_in[8];
  const float* g2   = (const float*)d_in[9];
  const float* be2  = (const float*)d_in[10];
  const float* Wih  = (const float*)d_in[11];
  const float* Whh  = (const float*)d_in[12];
  const float* bih  = (const float*)d_in[13];
  const float* bhh  = (const float*)d_in[14];
  const float* Wout = (const float*)d_in[15];
  const float* bout = (const float*)d_in[16];
  (void)in_sizes; (void)n_in; (void)out_size; (void)ws_size;

  __bf16* h_ws = (__bf16*)d_ws;                         // [B*T, H] bf16 (134 MB)
  float*  outp = (float*)d_out;                         // [B, T, OUT_DIM]
  float*  hxf  = outp + (size_t)Bsz * Tlen * OUT_DIM;   // [B, H]

  // 524288 tokens / (4 waves * 4 tiles * 16 rows) = 2048 blocks
  mlp_fused_kernel<<<2048, 128, 0, stream>>>(x, W1, b1, g1, be1, W2, b2, g2, be2, h_ws);
  // 1024 batch rows / 16-row WMMA tile = 64 independent scan blocks (1 wave each)
  gru_scan_kernel<<<64, 32, 0, stream>>>(isin, hx, Wih, Whh, bih, bhh, Wout, bout,
                                         h_ws, outp, hxf);
}
